// SelfAttention_31336081392236
// MI455X (gfx1250) — matrix operand order repaired and checked by reading, not compiled, not run on hardware
//
#include <hip/hip_runtime.h>
#include <hip/hip_bf16.h>

// ---------------------------------------------------------------- constants
constexpr int N_ = 4;
constexpr int L_ = 2048;
constexpr int H_ = 16;
constexpr int D_ = 64;
constexpr int E_ = 1024;  // H_*D_

typedef _Float16 v16h __attribute__((ext_vector_type(16)));
typedef _Float16 v8h  __attribute__((ext_vector_type(8)));
typedef float    v8f  __attribute__((ext_vector_type(8)));
typedef float    v4f  __attribute__((ext_vector_type(4)));
typedef unsigned u32x4 __attribute__((ext_vector_type(4)));
typedef int      i32x4 __attribute__((ext_vector_type(4)));
typedef int      i32x8 __attribute__((ext_vector_type(8)));

#define DEV __device__ __forceinline__

#if defined(__has_builtin)
#if __has_builtin(__builtin_amdgcn_tensor_load_to_lds) && \
    __has_builtin(__builtin_amdgcn_s_wait_tensorcnt)
#define HAVE_TDM 1
#endif
#endif
#ifndef HAVE_TDM
#define HAVE_TDM 0
#endif

// D = A(16x32 f16) * B(32x16 f16) + C(16x16 f32)
DEV v8f wmma_f16(v16h a, v16h b, v8f c) {
  return __builtin_amdgcn_wmma_f32_16x16x32_f16(false, a, false, b, (short)0, c,
                                                false, false);
}

// A-fragment (16x32, f16) from row-major halves. `p` = this lane's row base
// (already offset to the 32-wide K chunk). Layout: lane-half hA holds
// K = {16*hA .. 16*hA+15} contiguous.
DEV v16h a_frag_h(const _Float16* p, int hA) {
  const _Float16* q = p + 16 * hA;
  v8h lo = *(const v8h*)(q);
  v8h hi = *(const v8h*)(q + 8);
  return __builtin_shufflevector(lo, hi, 0, 1, 2, 3, 4, 5, 6, 7, 8, 9, 10, 11,
                                 12, 13, 14, 15);
}

// Same but converting from f32 rows (for the raw inputs).
DEV v16h a_frag_f(const float* p, int hA) {
  const float* q = p + 16 * hA;
  v16h a;
#pragma unroll
  for (int i = 0; i < 8; ++i) a[i] = (_Float16)q[i];
#pragma unroll
  for (int i = 0; i < 8; ++i) a[i + 8] = (_Float16)q[8 + i];
  return a;
}

// Cross-lane helpers (wave32): XOR-16 partner exchange and arbitrary broadcast.
DEV float xor16(float v) {
  return __builtin_bit_cast(
      float, __builtin_amdgcn_ds_swizzle(__builtin_bit_cast(int, v), 0x401F));
}
DEV float lane_bcast(float v, int src) {
  return __builtin_bit_cast(
      float, __builtin_amdgcn_ds_bpermute(src << 2, __builtin_bit_cast(int, v)));
}

// Wave-relative LDS byte offset of a shared-memory pointer: the flat shared
// address is {aperture_hi32, lds_offset32}, so truncation gives the offset.
DEV unsigned lds_off(const void* p) {
  return (unsigned)(unsigned long long)p;
}

#if HAVE_TDM
// Issue a TDM 2D tile load Global->LDS (D# built per ISA ch.8).
// tile_d0 = contiguous elements per row, tile_d1 = rows, stride0 = row stride
// (all in 2-byte elements). Tile rows land contiguously in LDS.
DEV void tdm_load_2d(unsigned ldsAddr, const void* gptr, unsigned tile_d0,
                     unsigned tile_d1, unsigned stride0) {
  unsigned long long ga = (unsigned long long)gptr;
  u32x4 g0;
  g0[0] = 1u;                                   // count=1, user descriptor
  g0[1] = ldsAddr;                              // lds_addr
  g0[2] = (unsigned)ga;                         // global_addr[31:0]
  g0[3] = (unsigned)((ga >> 32) & 0x01ffffffu)  // global_addr[56:32]
          | (2u << 30);                         // type=2 ("image")
  i32x8 g1;
  g1[0] = (int)(1u << 16);                          // data_size=1 -> 2 bytes
  g1[1] = (int)(tile_d0 << 16);                     // tensor_dim0 = tile_d0
  g1[2] = (int)((tile_d0 >> 16) | (tile_d1 << 16)); // tensor_dim1 = tile_d1
  g1[3] = (int)((tile_d1 >> 16) | (tile_d0 << 16)); // tile_dim0
  g1[4] = (int)tile_d1;                             // tile_dim1 (tile_dim2=0)
  g1[5] = (int)stride0;                             // tensor_dim0_stride lo
  g1[6] = 0;                                        // stride hi / dim1_stride
  g1[7] = 0;
  i32x4 gz = {0, 0, 0, 0};  // 2D tile: groups 2/3 unused (dims/tiles = 0)
#if __clang_major__ >= 23
  i32x8 gz8 = {0, 0, 0, 0, 0, 0, 0, 0};
  __builtin_amdgcn_tensor_load_to_lds(g0, g1, gz, gz, gz8, 0);
#else
  __builtin_amdgcn_tensor_load_to_lds(g0, g1, gz, gz, 0);
#endif
}
#endif

// ---------------------------------------------------------------- Wo -> f16
__global__ __launch_bounds__(256) void cvt_wo(const float* __restrict__ Wo,
                                              _Float16* __restrict__ Woh) {
  int i = (blockIdx.x * 256 + threadIdx.x) * 4;
  v4f f = *(const v4f*)(Wo + i);
#pragma unroll
  for (int j = 0; j < 4; ++j) Woh[i + j] = (_Float16)f[j];
}

// ------------------------------------------------- QKV projection (f16 out)
// grid: (N*H, L/128, 3), block 256 (8 waves, 16 rows each).
// out = X @ W^T per head; Q/K stored [n][h][l][d], V stored [n][h][d][l].
__global__ __launch_bounds__(256) void qkv_proj(
    const float* __restrict__ query, const float* __restrict__ key_,
    const float* __restrict__ value, const float* __restrict__ Wq,
    const float* __restrict__ Wk, const float* __restrict__ Wv,
    _Float16* __restrict__ Qh, _Float16* __restrict__ Kh,
    _Float16* __restrict__ Vt) {
  __shared__ alignas(32) _Float16 ldsW[64 * 64];
  const int nh = blockIdx.x;
  const int n = nh >> 4, h = nh & 15;
  const int z = blockIdx.z;
  const float* X = (z == 0) ? query : (z == 1) ? key_ : value;
  const float* W = (z == 0) ? Wq : (z == 1) ? Wk : Wv;

  const int tid = threadIdx.x;
#pragma unroll
  for (int i = 0; i < 16; ++i) {
    int idx = i * 256 + tid;
    ldsW[idx] = (_Float16)W[idx];
  }
  __syncthreads();

  const int wave = tid >> 5, lane = tid & 31;
  const int m = lane & 15, hA = lane >> 4;
  const int lrow = blockIdx.y * 128 + wave * 16;

  // A fragments: this lane's row l = lrow + m, K = d in [0,64)
  const float* rowp = X + ((size_t)(n * L_ + (lrow + m)) * H_ + h) * D_;
  v16h a0 = a_frag_f(rowp, hA);
  v16h a1 = a_frag_f(rowp + 32, hA);

#pragma unroll
  for (int es = 0; es < 4; ++es) {
    // B: lane column e = es*16 + m holds W[e, d] contiguous in d
    const _Float16* bp = ldsW + (es * 16 + m) * 64 + 16 * hA;
    v16h b0 = *(const v16h*)(bp);
    v16h b1 = *(const v16h*)(bp + 32);
    v8f c = {};
    c = wmma_f16(a0, b0, c);
    c = wmma_f16(a1, b1, c);

    if (z == 2) {  // V: transposed store [n][h][e][l], 8 contiguous l per lane
      v8h o;
#pragma unroll
      for (int j = 0; j < 8; ++j) o[j] = (_Float16)c[j];
      _Float16* dst =
          Vt + ((size_t)(nh * 64 + es * 16 + m)) * L_ + lrow + 8 * hA;
      *(v8h*)dst = o;
    } else {  // Q/K: [n][h][l][e]
      _Float16* dstT = ((z == 0) ? Qh : Kh) + (size_t)nh * L_ * 64;
      const int e = es * 16 + m;
#pragma unroll
      for (int j = 0; j < 8; ++j)
        dstT[(size_t)(lrow + 8 * hA + j) * 64 + e] = (_Float16)c[j];
    }
  }
}

// ----------------------------------------------------- flash attention core
// grid: (N*H, L/128), block 256 (8 waves, 16 queries each).
// Computes S^T = K Q^T per 32-key block so per-query softmax reductions are
// in-lane + one XOR-16 exchange, and P (C-layout) maps in-lane to the
// A-fragment of P^T for the P^T·V WMMA. K/V^T tiles are staged into LDS by
// the Tensor Data Mover (wave 0 issues, TENSORcnt + barrier publishes).
__global__ __launch_bounds__(256) void attention(const _Float16* __restrict__ Qh,
                                                 const _Float16* __restrict__ Kh,
                                                 const _Float16* __restrict__ Vt,
                                                 _Float16* __restrict__ Xh) {
  __shared__ alignas(32) _Float16 ldsK[32 * 64];  // [key][d]
  __shared__ alignas(32) _Float16 ldsV[64 * 32];  // [d][key]
  const int nh = blockIdx.x;
  const int n = nh >> 4, h = nh & 15;
  const int tid = threadIdx.x, wave = tid >> 5, lane = tid & 31;
  const int m = lane & 15, hA = lane >> 4;
  const int qbase = blockIdx.y * 128 + wave * 16;

  const _Float16* Qbase = Qh + (size_t)nh * L_ * 64;
  const _Float16* Kbase = Kh + (size_t)nh * L_ * 64;
  const _Float16* Vbase = Vt + (size_t)nh * 64 * L_;

  // Q^T B-fragments: lane column q = m holds Q[q, d] contiguous
  const _Float16* qp = Qbase + (size_t)(qbase + m) * 64 + 16 * hA;
  v16h bq0 = *(const v16h*)(qp);
  v16h bq1 = *(const v16h*)(qp + 32);

  float mrun = -1e30f, srun = 0.f;
  v8f o0 = {}, o1 = {}, o2 = {}, o3 = {};
  const float csc = 0.0450842200f;  // log2(e) / sqrt(EMBED=1024)

  const int dr = tid >> 2, cg = (tid & 3) * 8;
  const unsigned ldsKoff = lds_off(ldsK), ldsVoff = lds_off(ldsV);
  (void)ldsKoff;
  (void)ldsVoff;

  for (int kb = 0; kb < L_; kb += 32) {
#if HAVE_TDM
    if (wave == 0) {  // TDM: K tile 32 rows x 64 contig, V^T tile 64 x 32
      tdm_load_2d(ldsKoff, Kbase + (size_t)kb * 64, 64u, 32u, 64u);
      tdm_load_2d(ldsVoff, Vbase + kb, 32u, 64u, (unsigned)L_);
      __builtin_amdgcn_s_wait_tensorcnt(0);
    }
#else
    *(v8h*)(ldsK + tid * 8) = *(const v8h*)(Kbase + (size_t)kb * 64 + tid * 8);
    *(v8h*)(ldsV + dr * 32 + cg) =
        *(const v8h*)(Vbase + (size_t)dr * L_ + kb + cg);
#endif
    if (kb + 32 < L_) {  // prefetch next tiles into L2 (global_prefetch)
      __builtin_prefetch(Kbase + (size_t)(kb + 32) * 64 + tid * 8, 0, 1);
      __builtin_prefetch(Vbase + (size_t)dr * L_ + kb + 32 + cg, 0, 1);
    }
    __syncthreads();

    // S^T (32 keys x 16 queries): A = K rows, B = Q^T
    v8f s0 = {}, s1 = {};
    {
      const _Float16* k0 = ldsK + m * 64;
      const _Float16* k1 = ldsK + (16 + m) * 64;
      s0 = wmma_f16(a_frag_h(k0, hA), bq0, s0);
      s0 = wmma_f16(a_frag_h(k0 + 32, hA), bq1, s0);
      s1 = wmma_f16(a_frag_h(k1, hA), bq0, s1);
      s1 = wmma_f16(a_frag_h(k1 + 32, hA), bq1, s1);
    }

    // online softmax for column q = m (this lane + its XOR-16 partner)
    float bm = -1e30f;
#pragma unroll
    for (int j = 0; j < 8; ++j) bm = fmaxf(bm, fmaxf(s0[j], s1[j]));
    bm = fmaxf(bm, xor16(bm));
    const float mnew = fmaxf(mrun, bm);
    const float alpha = exp2f((mrun - mnew) * csc);
    mrun = mnew;

    v16h ap;  // P^T A-fragment, built fully in-lane
    float bsum = 0.f;
#pragma unroll
    for (int j = 0; j < 8; ++j) {
      float p0 = exp2f((s0[j] - mnew) * csc);
      float p1 = exp2f((s1[j] - mnew) * csc);
      bsum += p0 + p1;
      ap[j] = (_Float16)p0;
      ap[j + 8] = (_Float16)p1;
    }
    bsum += xor16(bsum);
    srun = srun * alpha + bsum;

    // rescale O rows (row q = j + 8*hA) by alpha[q]
#pragma unroll
    for (int j = 0; j < 8; ++j) {
      float aj = lane_bcast(alpha, j + 8 * hA);
      o0[j] *= aj;
      o1[j] *= aj;
      o2[j] *= aj;
      o3[j] *= aj;
    }

    // O += P^T · V  (B: lane column d = ds*16 + m, K = keys {8*hA..+7, 16+8*hA..+7})
    const _Float16* vb = ldsV + m * 32 + 8 * hA;
    auto vfrag = [](const _Float16* p) {
      v8h lo = *(const v8h*)(p);
      v8h hi = *(const v8h*)(p + 16);
      return __builtin_shufflevector(lo, hi, 0, 1, 2, 3, 4, 5, 6, 7, 8, 9, 10,
                                     11, 12, 13, 14, 15);
    };
    o0 = wmma_f16(ap, vfrag(vb), o0);
    o1 = wmma_f16(ap, vfrag(vb + 16 * 32), o1);
    o2 = wmma_f16(ap, vfrag(vb + 32 * 32), o2);
    o3 = wmma_f16(ap, vfrag(vb + 48 * 32), o3);
    __syncthreads();
  }

  const float inv = 1.f / srun;
  _Float16* xb = Xh + ((size_t)(n * L_ + qbase)) * E_ + h * 64 + m;
#pragma unroll
  for (int j = 0; j < 8; ++j) {
    float rj = lane_bcast(inv, j + 8 * hA);
    size_t r = (size_t)(j + 8 * hA) * E_;
    xb[r] = (_Float16)(o0[j] * rj);
    xb[r + 16] = (_Float16)(o1[j] * rj);
    xb[r + 32] = (_Float16)(o2[j] * rj);
    xb[r + 48] = (_Float16)(o3[j] * rj);
  }
}

// ------------------------------------------------- output projection + bias
// grid: (N*L/128, E/64), block 256. out = Xh @ Wo^T + bo, f32 result.
__global__ __launch_bounds__(256) void out_proj(const _Float16* __restrict__ Xh,
                                                const _Float16* __restrict__ Woh,
                                                const float* __restrict__ bo,
                                                float* __restrict__ out) {
  __shared__ alignas(32) _Float16 ldsW[64 * 32];  // [e][c] tile
  const int tid = threadIdx.x, wave = tid >> 5, lane = tid & 31;
  const int m = lane & 15, hA = lane >> 4;
  const int rbase = blockIdx.x * 128 + wave * 16;
  const int ebase = blockIdx.y * 64;
  const int er = tid >> 2, cg = (tid & 3) * 8;

  v8f c0 = {}, c1 = {}, c2 = {}, c3 = {};
  for (int kc = 0; kc < E_; kc += 32) {
    *(v8h*)(ldsW + er * 32 + cg) =
        *(const v8h*)(Woh + (size_t)(ebase + er) * E_ + kc + cg);
    __syncthreads();

    v16h a = a_frag_h(Xh + (size_t)(rbase + m) * E_ + kc, hA);
    const _Float16* bp = ldsW + m * 32 + 16 * hA;
    c0 = wmma_f16(a, *(const v16h*)(bp), c0);
    c1 = wmma_f16(a, *(const v16h*)(bp + 512), c1);
    c2 = wmma_f16(a, *(const v16h*)(bp + 1024), c2);
    c3 = wmma_f16(a, *(const v16h*)(bp + 1536), c3);
    __syncthreads();
  }

  const float b0 = bo[ebase + m], b1 = bo[ebase + 16 + m];
  const float b2 = bo[ebase + 32 + m], b3 = bo[ebase + 48 + m];
  float* ob = out + (size_t)rbase * E_ + ebase + m;
#pragma unroll
  for (int j = 0; j < 8; ++j) {
    size_t r = (size_t)(j + 8 * hA) * E_;
    ob[r] = c0[j] + b0;
    ob[r + 16] = c1[j] + b1;
    ob[r + 32] = c2[j] + b2;
    ob[r + 48] = c3[j] + b3;
  }
}

// ---------------------------------------------------------------- launcher
extern "C" void kernel_launch(void* const* d_in, const int* in_sizes, int n_in,
                              void* d_out, int out_size, void* d_ws,
                              size_t ws_size, hipStream_t stream) {
  const float* query = (const float*)d_in[0];
  const float* key_ = (const float*)d_in[1];
  const float* value = (const float*)d_in[2];
  const float* Wq = (const float*)d_in[3];
  const float* Wk = (const float*)d_in[4];
  const float* Wv = (const float*)d_in[5];
  const float* Wo = (const float*)d_in[6];
  const float* bo = (const float*)d_in[7];

  const size_t nTok = (size_t)N_ * L_ * E_;  // 8,388,608 elements
  _Float16* Qh = (_Float16*)d_ws;
  _Float16* Kh = Qh + nTok;
  _Float16* Vt = Kh + nTok;
  _Float16* Xh = Vt + nTok;
  _Float16* Woh = Xh + nTok;  // +2MB, total ws use ~66MB

  cvt_wo<<<(E_ * E_) / (256 * 4), 256, 0, stream>>>(Wo, Woh);
  qkv_proj<<<dim3(N_ * H_, L_ / 128, 3), 256, 0, stream>>>(
      query, key_, value, Wq, Wk, Wv, Qh, Kh, Vt);
  attention<<<dim3(N_ * H_, L_ / 128), 256, 0, stream>>>(Qh, Kh, Vt, Xh);
  out_proj<<<dim3((N_ * L_) / 128, E_ / 64), 256, 0, stream>>>(Xh, Woh, bo,
                                                               (float*)d_out);
}
